// LinearGM_7206955122812
// MI455X (gfx1250) — 8-bit matrix operand order repaired and checked by reading, compiled, not run on hardware
//
#include <hip/hip_runtime.h>

#define DIM_T 8192
#define DIM_K 4096
#define DIM_O 4096
#define KBLK  128
#define NKB   (DIM_K / KBLK)
#define FP8_MAX 448.0f

#define BM 128
#define BN 128
#define BSTR 144

typedef __attribute__((ext_vector_type(16))) int   v16i;
typedef __attribute__((ext_vector_type(8)))  float v8f;

__global__ __launch_bounds__(256) void quant_x_kernel(
    const float* __restrict__ x,
    unsigned char* __restrict__ xq,
    float* __restrict__ sa)
{
    const int wave = (int)((blockIdx.x * blockDim.x + threadIdx.x) >> 5);
    const int lane = threadIdx.x & 31;
    const int row  = wave >> 5;
    const int kb   = wave & 31;

    const float4 v = ((const float4*)(x + (size_t)row * DIM_K + kb * KBLK))[lane];
    float amax = fmaxf(fmaxf(fabsf(v.x), fabsf(v.y)),
                       fmaxf(fabsf(v.z), fabsf(v.w)));
#pragma unroll
    for (int m = 16; m >= 1; m >>= 1)
        amax = fmaxf(amax, __shfl_xor(amax, m, 32));
    amax = fmaxf(amax, 1e-12f);

    const float inv = FP8_MAX / amax;
    int packed = 0;
    packed = __builtin_amdgcn_cvt_pk_fp8_f32(v.x * inv, v.y * inv, packed, false);
    packed = __builtin_amdgcn_cvt_pk_fp8_f32(v.z * inv, v.w * inv, packed, true);
    *(int*)(xq + (size_t)row * DIM_K + kb * KBLK + lane * 4) = packed;

    if (lane == 0) sa[row * NKB + kb] = amax * (1.0f / FP8_MAX);
}

__global__ __launch_bounds__(256) void quant_w_kernel(
    const float* __restrict__ w,
    unsigned char* __restrict__ wq,
    float* __restrict__ sb)
{
    __shared__ float red[256];
    const int tile = blockIdx.x;
    const int tr = tile >> 5, tc = tile & 31;
    const int t   = threadIdx.x;
    const int row = t >> 1;
    const int ch  = t & 1;

    const float4* base = (const float4*)(w + (size_t)(tr * 128 + row) * DIM_K
                                           + tc * 128 + ch * 64);
    float4 v[16];
    float amax = 0.0f;
#pragma unroll
    for (int i = 0; i < 16; ++i) {
        v[i] = base[i];
        amax = fmaxf(amax, fmaxf(fmaxf(fabsf(v[i].x), fabsf(v[i].y)),
                                 fmaxf(fabsf(v[i].z), fabsf(v[i].w))));
    }
    red[t] = amax;
    __syncthreads();
#pragma unroll
    for (int s = 128; s >= 1; s >>= 1) {
        if (t < s) red[t] = fmaxf(red[t], red[t + s]);
        __syncthreads();
    }
    const float tamax = fmaxf(red[0], 1e-12f);
    const float inv   = FP8_MAX / tamax;

    int pk[16];
#pragma unroll
    for (int i = 0; i < 16; ++i) {
        int p = __builtin_amdgcn_cvt_pk_fp8_f32(v[i].x * inv, v[i].y * inv, 0, false);
        pk[i] = __builtin_amdgcn_cvt_pk_fp8_f32(v[i].z * inv, v[i].w * inv, p, true);
    }
    int4* ob = (int4*)(wq + (size_t)(tr * 128 + row) * DIM_K + tc * 128 + ch * 64);
#pragma unroll
    for (int i = 0; i < 4; ++i)
        ob[i] = make_int4(pk[4 * i], pk[4 * i + 1], pk[4 * i + 2], pk[4 * i + 3]);

    if (t == 0) sb[tr * NKB + tc] = tamax * (1.0f / FP8_MAX);
}

__global__ __launch_bounds__(256) void gemm_fp8_kernel(
    const unsigned char* __restrict__ xq,
    const unsigned char* __restrict__ wq,
    const float* __restrict__ sa,
    const float* __restrict__ sb,
    const float* __restrict__ bias,
    float* __restrict__ out)
{
    __shared__ unsigned char Bs[2][BM * BSTR];

    const int t    = threadIdx.x;
    const int w    = t >> 5;
    const int lane = t & 31;
    const int l15  = lane & 15;
    const int hi   = lane >> 4;

    const int bm = (int)blockIdx.x & 63;
    const int bn = (int)blockIdx.x >> 6;
    const int m0 = bm * BM;
    const int n0 = bn * BN;
    const int mrow = m0 + w * 16 + l15;

    const unsigned ldsB0 = (unsigned)(size_t)&Bs[0][0];
    const unsigned ldsB1 = (unsigned)(size_t)&Bs[1][0];

    auto stage_b = [&](int kb, unsigned ldsbase) {
#pragma unroll
        for (int i = 0; i < 4; ++i) {
            const int c   = t + i * 256;
            const int row = c >> 3;
            const int o16 = c & 7;
            const unsigned loff = ldsbase + (unsigned)(row * BSTR + o16 * 16);
            const unsigned char* g =
                wq + (size_t)(n0 + row) * DIM_K + kb * KBLK + o16 * 16;
            asm volatile("global_load_async_to_lds_b128 %0, %1, off"
                         :: "v"(loff), "v"(g) : "memory");
        }
    };

    v8f acc[8];
#pragma unroll
    for (int j = 0; j < 8; ++j)
#pragma unroll
        for (int v = 0; v < 8; ++v) acc[j][v] = 0.0f;

    stage_b(0, ldsB0);

    for (int kb = 0; kb < NKB; ++kb) {
        if (kb + 1 < NKB) {
            stage_b(kb + 1, ((kb + 1) & 1) ? ldsB1 : ldsB0);
            asm volatile("s_wait_asynccnt 0x4" ::: "memory");
        } else {
            asm volatile("s_wait_asynccnt 0x0" ::: "memory");
        }
        __syncthreads();

        v16i A;
        const unsigned char* ab = xq + (size_t)mrow * DIM_K + kb * KBLK + hi * 16;
#pragma unroll
        for (int c = 0; c < 8; ++c) {
            int2 d = *(const int2*)(ab + (c >> 1) * 32 + (c & 1) * 8);
            A[2 * c]     = d.x;
            A[2 * c + 1] = d.y;
        }

        const float sbk = sb[bn * NKB + kb];
        float s[8];
#pragma unroll
        for (int v = 0; v < 8; ++v)
            s[v] = sa[(m0 + w * 16 + v + hi * 8) * NKB + kb] * sbk;

        const unsigned char* bbase = &Bs[kb & 1][0];
#pragma unroll
        for (int j = 0; j < 8; ++j) {
            const unsigned char* bp = bbase + (j * 16 + l15) * BSTR + hi * 16;
            v16i B;
#pragma unroll
            for (int c = 0; c < 4; ++c) {
                int4 d = *(const int4*)(bp + c * 32);
                B[4 * c]     = d.x;
                B[4 * c + 1] = d.y;
                B[4 * c + 2] = d.z;
                B[4 * c + 3] = d.w;
            }
            v8f z = {0.f, 0.f, 0.f, 0.f, 0.f, 0.f, 0.f, 0.f};
            v8f r = __builtin_amdgcn_wmma_f32_16x16x128_fp8_fp8(
                A, B, (short)0, z, false, false);
#pragma unroll
            for (int v = 0; v < 8; ++v)
                acc[j][v] += r[v] * s[v];
        }
        __syncthreads();
    }

#pragma unroll
    for (int j = 0; j < 8; ++j) {
        const int ncol = n0 + j * 16 + l15;
        const float bv = bias[ncol];
#pragma unroll
        for (int v = 0; v < 8; ++v) {
            const int row = m0 + w * 16 + v + hi * 8;
            out[(size_t)row * DIM_O + ncol] = acc[j][v] + bv;
        }
    }
}

extern "C" void kernel_launch(void* const* d_in, const int* in_sizes, int n_in,
                              void* d_out, int out_size, void* d_ws, size_t ws_size,
                              hipStream_t stream) {
    const float* x    = (const float*)d_in[0];
    const float* w    = (const float*)d_in[1];
    const float* bias = (const float*)d_in[2];
    float* out        = (float*)d_out;

    unsigned char* ws = (unsigned char*)d_ws;
    unsigned char* xq = ws;
    unsigned char* wq = xq + (size_t)DIM_T * DIM_K;
    float* sa = (float*)(wq + (size_t)DIM_O * DIM_K);
    float* sb = sa + (size_t)DIM_T * NKB;

    {
        const int waves = DIM_T * NKB;
        quant_x_kernel<<<waves / 8, 256, 0, stream>>>(x, xq, sa);
    }
    {
        const int tiles = (DIM_O / 128) * (DIM_K / 128);
        quant_w_kernel<<<tiles, 256, 0, stream>>>(w, wq, sb);
    }
    {
        const int blocks = (DIM_T / BM) * (DIM_O / BN);
        gemm_fp8_kernel<<<blocks, 256, 0, stream>>>(xq, wq, sa, sb, bias, out);
    }
}
